// SwinTransformerBlock_revised_54254026883480
// MI455X (gfx1250) — matrix operand order repaired and checked by reading, not compiled, not run on hardware
//
#include <hip/hip_runtime.h>
#include <hip/hip_bf16.h>
#include <math.h>

typedef __attribute__((ext_vector_type(16))) _Float16 v16h;
typedef __attribute__((ext_vector_type(8)))  float    v8f;

#define DIMC 96
#define HRES 128

union FragU { v16h v; uint4 q[2]; };

// A-matrix 16x32 f16 fragment from row-major LDS tile (ld in halves).
// ISA layout: lane(m = lane&15, g = lane>>4); VGPR0-3: K = g*8+0..7; VGPR4-7: K = 16+g*8+0..7
__device__ __forceinline__ v16h frag_a(const _Float16* base, int ld) {
  const int lane = threadIdx.x & 31;
  const int g = lane >> 4, m = lane & 15;
  FragU u;
  const _Float16* p = base + m * ld + g * 8;
  u.q[0] = *(const uint4*)(p);
  u.q[1] = *(const uint4*)(p + 16);
  return u.v;
}

// B-matrix 32x16 f16 fragment from N-major (transposed) LDS copy Bt[n][k] (ld in halves).
// ISA layout: lane(n = lane&15, g = lane>>4); VGPR0-3: K = g*8+0..7; VGPR4-7: K = 16+g*8+0..7
__device__ __forceinline__ v16h frag_bt(const _Float16* bt, int ld) {
  const int lane = threadIdx.x & 31;
  const int g = lane >> 4, n = lane & 15;
  FragU u;
  const _Float16* p = bt + n * ld + g * 8;
  u.q[0] = *(const uint4*)(p);
  u.q[1] = *(const uint4*)(p + 16);
  return u.v;
}

__device__ __forceinline__ v8f wmma16(v16h a, v16h b, v8f c) {
  return __builtin_amdgcn_wmma_f32_16x16x32_f16(false, a, false, b, (short)0, c, false, false);
}

// ---------------------------------------------------------------------------
// Kernel 0: CPB-MLP relative position bias -> rpb[3][64][64]
// ---------------------------------------------------------------------------
extern "C" __global__ __launch_bounds__(256)
void swin_rpb_kernel(const float* __restrict__ cw1, const float* __restrict__ cb1,
                     const float* __restrict__ cw2, float* __restrict__ rpb) {
  __shared__ float BT[225 * 3];
  const int tid = threadIdx.x;
  if (tid < 225) {
    int ii = tid / 15, jj = tid % 15;
    float c0 = (float)(ii - 7) * (8.0f / 7.0f);
    float c1 = (float)(jj - 7) * (8.0f / 7.0f);
    float t0 = copysignf(log2f(fabsf(c0) + 1.0f) * (1.0f / 3.0f), c0);
    float t1 = copysignf(log2f(fabsf(c1) + 1.0f) * (1.0f / 3.0f), c1);
    float a0 = 0.f, a1 = 0.f, a2 = 0.f;
    for (int j = 0; j < 512; ++j) {
      float hv = fmaxf(t0 * cw1[j * 2] + t1 * cw1[j * 2 + 1] + cb1[j], 0.0f);
      a0 += hv * cw2[j];
      a1 += hv * cw2[512 + j];
      a2 += hv * cw2[1024 + j];
    }
    BT[tid * 3 + 0] = a0; BT[tid * 3 + 1] = a1; BT[tid * 3 + 2] = a2;
  }
  __syncthreads();
  for (int i = tid; i < 3 * 64 * 64; i += 256) {
    int h = i >> 12, rem = i & 4095, nq = rem >> 6, mk = rem & 63;
    int dh = (nq >> 3) - (mk >> 3) + 7;
    int dw = (nq & 7) - (mk & 7) + 7;
    float v = BT[(dh * 15 + dw) * 3 + h];
    rpb[i] = 16.0f / (1.0f + expf(-v));
  }
}

// ---------------------------------------------------------------------------
// Kernel 1: fused window attention (shift-gather, qkv, cosine attn, softmax,
// P*V, projection). One workgroup (8 wave32) per 8x8 window.
// ---------------------------------------------------------------------------
extern "C" __global__ __launch_bounds__(256)
void swin_attn_kernel(const float* __restrict__ x,
                      const float* __restrict__ qkv_w,
                      const float* __restrict__ q_bias,
                      const float* __restrict__ v_bias,
                      const float* __restrict__ logit_scale,
                      const float* __restrict__ proj_w,
                      const float* __restrict__ proj_b,
                      const float* __restrict__ rpb,
                      float* __restrict__ y) {
  extern __shared__ char smem[];
  _Float16* W16 = (_Float16*)(smem + 0);       // [288][96] qkv_w (N-major == row-major of qkv_w)
  float*    S   = (float*)   (smem + 0);       // [3][64][64]  (aliases W16 after qkv phase)
  _Float16* XW  = (_Float16*)(smem + 55296);   // [64][96] tokens f16
  float*    QF  = (float*)   (smem + 67584);   // [3][64][32] q f32
  _Float16* P16 = (_Float16*)(smem + 67584);   // [3][64][64]  (aliases QF after norm)
  float*    KF  = (float*)   (smem + 92160);   // [3][64][32] k f32
  _Float16* O16 = (_Float16*)(smem + 92160);   // [64][96]     (aliases KF after norm)
  _Float16* PW  = (_Float16*)(smem + 116736);  // [96][96] proj_w
  _Float16* Q16 = (_Float16*)(smem + 135168);  // [3][64][32] normalized q
  _Float16* K16 = (_Float16*)(smem + 147456);  // [3][64][32] normalized k
  _Float16* VT  = (_Float16*)(smem + 159744);  // [3][32][64] v transposed (N-major)
  int*      REG = (int*)     (smem + 172032);  // [64] shift-mask region ids

  const int tid  = threadIdx.x;
  const int blk  = blockIdx.x;
  const int b    = blk >> 8;
  const int widx = blk & 255;
  const int wh = widx >> 4, ww = widx & 15;
  const long xbase = (long)b * (HRES * HRES) * DIMC;

  // ---- stage: gather shifted window + convert weights to f16 ----
  for (int i = tid; i < 64 * DIMC; i += 256) {
    int r = i / DIMC, c = i - r * DIMC;
    int sh = (wh * 8 + (r >> 3) + 4) & 127;   // roll(-4)
    int sw = (ww * 8 + (r & 7) + 4) & 127;
    XW[i] = (_Float16)x[xbase + (long)(sh * HRES + sw) * DIMC + c];
  }
  for (int i = tid; i < 288 * DIMC; i += 256) W16[i] = (_Float16)qkv_w[i];
  for (int i = tid; i < DIMC * DIMC; i += 256) PW[i] = (_Float16)proj_w[i];
  if (tid < 64) {
    int ha = wh * 8 + (tid >> 3), wa = ww * 8 + (tid & 7);
    int rh = (ha < 120) ? 0 : ((ha < 124) ? 1 : 2);
    int rw = (wa < 120) ? 0 : ((wa < 124) ? 1 : 2);
    REG[tid] = rh * 3 + rw;
  }
  __syncthreads();

  const int wv = tid >> 5;
  const int ln = tid & 15, lg = (tid & 31) >> 4;

  // ---- qkv = XW @ qkv_w^T + bias : M=64, N=288, K=96 (72 tiles / 8 waves) ----
  for (int t = wv; t < 72; t += 8) {
    int mt = t / 18, nt = t % 18;
    // segment (q/k/v) and head are wave-uniform per tile: boundaries are
    // 96- and 32-aligned, tile columns are 16-aligned.
    int col0 = nt * 16;
    int seg  = col0 / 96;           // uniform
    int rem0 = col0 - seg * 96;     // uniform
    int h    = rem0 >> 5;           // uniform
    int d    = (rem0 & 31) + ln;    // per-lane offset within head
    int o    = col0 + ln;
    float bias = (seg == 0) ? q_bias[o] : ((seg == 1) ? 0.0f : v_bias[o - 192]);
    v8f acc;
    #pragma unroll
    for (int r = 0; r < 8; ++r) acc[r] = bias;
    #pragma unroll
    for (int kc = 0; kc < 3; ++kc) {
      v16h a  = frag_a (XW  + mt * 16 * DIMC + kc * 32, DIMC);
      v16h bb = frag_bt(W16 + nt * 16 * DIMC + kc * 32, DIMC);
      acc = wmma16(a, bb, acc);
    }
    if (seg == 0) {
      #pragma unroll
      for (int r = 0; r < 8; ++r) {
        int m = mt * 16 + r + 8 * lg;
        QF[(h * 64 + m) * 32 + d] = acc[r];
      }
    } else if (seg == 1) {
      #pragma unroll
      for (int r = 0; r < 8; ++r) {
        int m = mt * 16 + r + 8 * lg;
        KF[(h * 64 + m) * 32 + d] = acc[r];
      }
    } else {
      #pragma unroll
      for (int r = 0; r < 8; ++r) {
        int m = mt * 16 + r + 8 * lg;
        VT[(h * 32 + d) * 64 + m] = (_Float16)acc[r];
      }
    }
  }
  __syncthreads();

  // ---- cosine-norm q,k rows (192 rows) ----
  if (tid < 192) {
    const float* q = QF + tid * 32;
    const float* k = KF + tid * 32;
    float sq = 0.f, sk = 0.f;
    for (int d = 0; d < 32; ++d) { sq += q[d] * q[d]; sk += k[d] * k[d]; }
    float rq = 1.0f / fmaxf(sqrtf(sq), 1e-12f);
    float rk = 1.0f / fmaxf(sqrtf(sk), 1e-12f);
    for (int d = 0; d < 32; ++d) {
      Q16[tid * 32 + d] = (_Float16)(q[d] * rq);
      K16[tid * 32 + d] = (_Float16)(k[d] * rk);
    }
  }
  __syncthreads();

  // ---- S = qn @ kn^T * ls + rpb + mask : 3 heads x 4x4 tiles ----
  for (int t = wv; t < 48; t += 8) {
    int h = t >> 4, r4 = t & 15;
    int mt = r4 >> 2, nt = r4 & 3;
    float ls = expf(fminf(logit_scale[h], 4.6051701860f)); // ln(100)
    v8f acc = {};
    v16h a  = frag_a (Q16 + h * 2048 + mt * 16 * 32, 32);
    v16h bb = frag_bt(K16 + h * 2048 + nt * 16 * 32, 32);
    acc = wmma16(a, bb, acc);
    #pragma unroll
    for (int r = 0; r < 8; ++r) {
      int row = mt * 16 + r + 8 * lg;  // query token
      int col = nt * 16 + ln;          // key token
      float v = acc[r] * ls + rpb[(h * 64 + row) * 64 + col]
              + ((REG[row] != REG[col]) ? -100.0f : 0.0f);
      S[(h * 64 + row) * 64 + col] = v;
    }
  }
  __syncthreads();

  // ---- softmax rows, write P in f16 ----
  if (tid < 192) {
    const float* srow = S + tid * 64;
    float mx = -1e30f;
    for (int j = 0; j < 64; ++j) mx = fmaxf(mx, srow[j]);
    float sum = 0.f;
    for (int j = 0; j < 64; ++j) sum += expf(srow[j] - mx);
    float rs = 1.0f / sum;
    _Float16* prow = P16 + tid * 64;
    for (int j = 0; j < 64; ++j) prow[j] = (_Float16)(expf(srow[j] - mx) * rs);
  }
  __syncthreads();

  // ---- O = P @ V : per head M=64,N=32,K=64 (24 tiles) ----
  for (int t = wv; t < 24; t += 8) {
    int h = t / 8, r2 = t % 8, mt = r2 >> 1, nt = r2 & 1;
    v8f acc = {};
    #pragma unroll
    for (int kc = 0; kc < 2; ++kc) {
      v16h a  = frag_a (P16 + h * 4096 + mt * 16 * 64 + kc * 32, 64);
      v16h bb = frag_bt(VT  + h * 2048 + nt * 16 * 64 + kc * 32, 64);
      acc = wmma16(a, bb, acc);
    }
    #pragma unroll
    for (int r = 0; r < 8; ++r) {
      int m = mt * 16 + r + 8 * lg;
      O16[m * DIMC + h * 32 + nt * 16 + ln] = (_Float16)acc[r];
    }
  }
  __syncthreads();

  // ---- Z = O @ proj_w^T + proj_b -> global y (window layout) ----
  for (int t = wv; t < 24; t += 8) {
    int mt = t / 6, nt = t % 6;
    float bias = proj_b[nt * 16 + ln];
    v8f acc;
    #pragma unroll
    for (int r = 0; r < 8; ++r) acc[r] = bias;
    #pragma unroll
    for (int kc = 0; kc < 3; ++kc) {
      v16h a  = frag_a (O16 + mt * 16 * DIMC + kc * 32, DIMC);
      v16h bb = frag_bt(PW  + nt * 16 * DIMC + kc * 32, DIMC);
      acc = wmma16(a, bb, acc);
    }
    #pragma unroll
    for (int r = 0; r < 8; ++r) {
      int m = mt * 16 + r + 8 * lg;
      y[((long)blk * 64 + m) * DIMC + nt * 16 + ln] = acc[r];
    }
  }
}

// ---------------------------------------------------------------------------
// Kernel 2: un-shift gather + residual + LN1, MLP (WMMA), residual + LN2
// One workgroup per 64 tokens.
// ---------------------------------------------------------------------------
extern "C" __global__ __launch_bounds__(256)
void swin_mlp_kernel(const float* __restrict__ x,
                     const float* __restrict__ n1w, const float* __restrict__ n1b,
                     const float* __restrict__ n2w, const float* __restrict__ n2b,
                     const float* __restrict__ w1,  const float* __restrict__ b1,
                     const float* __restrict__ w2,  const float* __restrict__ b2,
                     const float* __restrict__ y,
                     float* __restrict__ out) {
  extern __shared__ char smem[];
  _Float16* W   = (_Float16*)(smem);            // [384][96] w1, then [96][384] w2
  _Float16* H16 = (_Float16*)(smem + 73728);    // [64][96]
  _Float16* HD  = (_Float16*)(smem + 86016);    // [64][384] gelu(hidden)
  float*    H32 = (float*)   (smem + 135168);   // [64][96]
  float*    Z2  = (float*)   (smem + 159744);   // [64][96]

  const int tid  = threadIdx.x;
  const int blk  = blockIdx.x;
  const int b    = blk >> 8;
  const int lblk = blk & 255;

  for (int i = tid; i < 384 * DIMC; i += 256) W[i] = (_Float16)w1[i];
  if (tid < 64) {
    int l  = lblk * 64 + tid;
    int ph = l >> 7, pw = l & 127;
    int hs = (ph + 124) & 127, ws2 = (pw + 124) & 127;   // inverse roll(+4)
    int widx = (hs >> 3) * 16 + (ws2 >> 3);
    int n    = (hs & 7) * 8 + (ws2 & 7);
    const float* yr = y + ((long)(b * 256 + widx) * 64 + n) * DIMC;
    const float* xr = x + ((long)b * (HRES * HRES) + l) * DIMC;
    float mu = 0.f;
    for (int c = 0; c < DIMC; ++c) mu += yr[c];
    mu *= (1.0f / DIMC);
    float var = 0.f;
    for (int c = 0; c < DIMC; ++c) { float d = yr[c] - mu; var += d * d; }
    var *= (1.0f / DIMC);
    float rstd = rsqrtf(var + 1e-5f);
    for (int c = 0; c < DIMC; ++c) {
      float h = xr[c] + (yr[c] - mu) * rstd * n1w[c] + n1b[c];
      H32[tid * DIMC + c] = h;
      H16[tid * DIMC + c] = (_Float16)h;
    }
  }
  __syncthreads();

  const int wv = tid >> 5;
  const int ln = tid & 15, lg = (tid & 31) >> 4;

  // ---- hidden = gelu(H @ w1^T + b1) : M=64, N=384, K=96 (96 tiles) ----
  for (int t = wv; t < 96; t += 8) {
    int mt = t / 24, nt = t % 24;
    float bias = b1[nt * 16 + ln];
    v8f acc;
    #pragma unroll
    for (int r = 0; r < 8; ++r) acc[r] = bias;
    #pragma unroll
    for (int kc = 0; kc < 3; ++kc) {
      v16h a  = frag_a (H16 + mt * 16 * DIMC + kc * 32, DIMC);
      v16h bb = frag_bt(W   + nt * 16 * DIMC + kc * 32, DIMC);
      acc = wmma16(a, bb, acc);
    }
    #pragma unroll
    for (int r = 0; r < 8; ++r) {
      int m = mt * 16 + r + 8 * lg;
      float v = acc[r];
      float g = 0.5f * v * (1.0f + erff(v * 0.70710678118f)); // exact gelu
      HD[m * 384 + nt * 16 + ln] = (_Float16)g;
    }
  }
  __syncthreads();

  for (int i = tid; i < DIMC * 384; i += 256) W[i] = (_Float16)w2[i];
  __syncthreads();

  // ---- z2 = hidden @ w2^T + b2 : M=64, N=96, K=384 (24 tiles) ----
  for (int t = wv; t < 24; t += 8) {
    int mt = t / 6, nt = t % 6;
    float bias = b2[nt * 16 + ln];
    v8f acc;
    #pragma unroll
    for (int r = 0; r < 8; ++r) acc[r] = bias;
    #pragma unroll
    for (int kc = 0; kc < 12; ++kc) {
      v16h a  = frag_a (HD + mt * 16 * 384 + kc * 32, 384);
      v16h bb = frag_bt(W  + nt * 16 * 384 + kc * 32, 384);
      acc = wmma16(a, bb, acc);
    }
    #pragma unroll
    for (int r = 0; r < 8; ++r) {
      int m = mt * 16 + r + 8 * lg;
      Z2[m * DIMC + nt * 16 + ln] = acc[r];
    }
  }
  __syncthreads();

  // ---- out = H + LN2(z2) ----
  if (tid < 64) {
    const float* zr = Z2 + tid * DIMC;
    float mu = 0.f;
    for (int c = 0; c < DIMC; ++c) mu += zr[c];
    mu *= (1.0f / DIMC);
    float var = 0.f;
    for (int c = 0; c < DIMC; ++c) { float d = zr[c] - mu; var += d * d; }
    var *= (1.0f / DIMC);
    float rstd = rsqrtf(var + 1e-5f);
    int l = lblk * 64 + tid;
    float* orow = out + ((long)b * (HRES * HRES) + l) * DIMC;
    for (int c = 0; c < DIMC; ++c)
      orow[c] = H32[tid * DIMC + c] + (zr[c] - mu) * rstd * n2w[c] + n2b[c];
  }
}

// ---------------------------------------------------------------------------
extern "C" void kernel_launch(void* const* d_in, const int* in_sizes, int n_in,
                              void* d_out, int out_size, void* d_ws, size_t ws_size,
                              hipStream_t stream) {
  const float* x    = (const float*)d_in[0];
  const float* n1w  = (const float*)d_in[1];
  const float* n1b  = (const float*)d_in[2];
  const float* qkvw = (const float*)d_in[3];
  const float* qb   = (const float*)d_in[4];
  const float* vb   = (const float*)d_in[5];
  const float* lsc  = (const float*)d_in[6];
  const float* cw1  = (const float*)d_in[7];
  const float* cb1  = (const float*)d_in[8];
  const float* cw2  = (const float*)d_in[9];
  const float* pw   = (const float*)d_in[10];
  const float* pb   = (const float*)d_in[11];
  const float* n2w  = (const float*)d_in[12];
  const float* n2b  = (const float*)d_in[13];
  const float* w1   = (const float*)d_in[14];
  const float* b1   = (const float*)d_in[15];
  const float* w2   = (const float*)d_in[16];
  const float* b2   = (const float*)d_in[17];
  float* out = (float*)d_out;

  float* rpb = (float*)d_ws;                          // 48 KB
  float* y   = (float*)((char*)d_ws + 64 * 1024);     // B*256*64*96 f32, window layout

  const int B = in_sizes[0] / (HRES * HRES * DIMC);
  const int nblk = B * 256;

  swin_rpb_kernel<<<1, 256, 0, stream>>>(cw1, cb1, cw2, rpb);
  swin_attn_kernel<<<nblk, 256, 172288, stream>>>(x, qkvw, qb, vb, lsc, pw, pb, rpb, y);
  swin_mlp_kernel<<<nblk, 256, 184320, stream>>>(x, n1w, n1b, n2w, n2b, w1, b1, w2, b2, y, out);
}
